// WeakAD_50826642981543
// MI455X (gfx1250) — matrix operand order repaired and checked by reading, not compiled, not run on hardware
//
#include <hip/hip_runtime.h>
#include <hip/hip_bf16.h>

#define NNODES 8192
#define NEDGE  262144
#define NSUB   128
#define KSUB   128
#define NNEG   4
#define N64    (NNODES * 64)

typedef __attribute__((ext_vector_type(16))) _Float16 v16h;
typedef __attribute__((ext_vector_type(8)))  _Float16 v8h;
typedef __attribute__((ext_vector_type(8)))  float    v8f;
typedef __attribute__((ext_vector_type(4)))  unsigned int u32x4;
typedef __attribute__((ext_vector_type(8)))  int          i32x8;
typedef __attribute__((ext_vector_type(4)))  int          i32x4;

// ---------------------------------------------------------------------------
// WMMA fragment layouts per CDNA5 ISA 7.12.2 (wave32, 16x16x32 f16):
// A (16x32 MxK): lanes 0-15 row M=lane, K base 0; lanes 16-31 K base 8;
//   halves 0..7 -> K=kb..kb+7, halves 8..15 -> K=kb+16..kb+23.
// B (32x16 KxN): col = lane&15; K order as for A (lanes 0-15 K base 0, lanes 16-31 K base 8).
// C/D (16x16 f32): m = j + 8*(lane>>4), n = lane&15.
// ---------------------------------------------------------------------------
__device__ __forceinline__ v16h load_afrag_f32(const float* __restrict__ A,
                                               int rowbase, int k0) {
  const int lane = threadIdx.x & 31;
  const float* p = A + (size_t)(rowbase + (lane & 15)) * 64 + k0 + ((lane >> 4) << 3);
  v16h a;
#pragma unroll
  for (int i = 0; i < 8; ++i) a[i] = (_Float16)p[i];
#pragma unroll
  for (int i = 0; i < 8; ++i) a[8 + i] = (_Float16)p[16 + i];
  return a;
}

// B fragment from LDS tile stored K-contiguous per column: T[c*64 + k].
__device__ __forceinline__ v16h load_bfrag_lds(const _Float16* __restrict__ T,
                                               int colbase, int k0) {
  const int lane = threadIdx.x & 31;
  const _Float16* p = T + (size_t)(colbase + (lane & 15)) * 64 + k0 + ((lane >> 4) << 3);
  v8h lo = *(const v8h*)(p);
  v8h hi = *(const v8h*)(p + 16);
  v16h b;
#pragma unroll
  for (int i = 0; i < 8; ++i) { b[i] = lo[i]; b[8 + i] = hi[i]; }
  return b;
}

// ---------------------------------------------------------------------------
// Generic [8192,64] @ [64,64] GEMM, optional bias + PReLU, optional B^T.
// W staged in LDS as f16, pre-transposed to Wt[c][k] so each B fragment is
// two contiguous ds_load_b128. One wave = 16-row strip, 4 col tiles, K=64.
// ---------------------------------------------------------------------------
__global__ __launch_bounds__(256) void gemm64_kernel(
    const float* __restrict__ A, const float* __restrict__ W,
    const float* __restrict__ bias, const float* __restrict__ slope,
    float* __restrict__ C, int transB, int act) {
  __shared__ __align__(16) _Float16 Wt[64 * 64];  // Wt[c][k]
  for (int i = threadIdx.x; i < 64 * 64; i += 256) {
    const int c = i >> 6, k = i & 63;
    Wt[i] = (_Float16)(transB ? W[(size_t)c * 64 + k] : W[(size_t)k * 64 + c]);
  }
  const int wave = threadIdx.x >> 5;
  const int lane = threadIdx.x & 31;
  const int rowbase = blockIdx.x * 128 + wave * 16;
  v16h a0 = load_afrag_f32(A, rowbase, 0);
  v16h a1 = load_afrag_f32(A, rowbase, 32);
  const float sl = slope ? *slope : 0.f;
  __syncthreads();
#pragma unroll
  for (int ct = 0; ct < 4; ++ct) {
    v8f acc = {};
    v16h b0 = load_bfrag_lds(Wt, ct * 16, 0);
    acc = __builtin_amdgcn_wmma_f32_16x16x32_f16(false, a0, false, b0,
                                                 (short)0, acc, false, false);
    v16h b1 = load_bfrag_lds(Wt, ct * 16, 32);
    acc = __builtin_amdgcn_wmma_f32_16x16x32_f16(false, a1, false, b1,
                                                 (short)0, acc, false, false);
    const int col = ct * 16 + (lane & 15);
    const float bv = bias ? bias[col] : 0.f;
#pragma unroll
    for (int j = 0; j < 8; ++j) {
      const int r = rowbase + j + ((lane >> 4) << 3);
      float v = acc[j] + bv;
      if (act) v = (v >= 0.f) ? v : sl * v;
      C[(size_t)r * 64 + col] = v;
    }
  }
}

// ---------------------------------------------------------------------------
// adj = sigmoid(V @ V^T), V = node_vec as f16 (1MB, L2 resident).
// Block = 128x128 tile. The 128x64 f16 column tile (16KB) is staged into LDS
// ONCE per block via the Tensor Data Mover (tensor_load_to_lds, TENSORcnt),
// removing the 8x redundant per-wave L2 reads; waves then build B fragments
// from LDS. Output path is store-bandwidth bound (256MB @ 23.3TB/s).
// ---------------------------------------------------------------------------
__device__ __forceinline__ void tdm_load_tile_128x64(const _Float16* gsrc,
                                                     _Float16* lds_dst) {
  // D# group 0: count=1 | lds_addr | global_addr[56:0] | type=2
  const unsigned long long ga = (unsigned long long)(uintptr_t)gsrc;
  const unsigned int laddr = (unsigned int)(uintptr_t)lds_dst;  // addr[31:0] = LDS offset
  u32x4 g0;
  g0[0] = 1u;                                   // count=1, is_restore=0, gather=0
  g0[1] = laddr;                                // lds_addr (bytes)
  g0[2] = (unsigned int)ga;                     // global_addr[31:0]
  g0[3] = (unsigned int)((ga >> 32) & 0x01FFFFFFu) | (2u << 30);  // [56:32] | type=2
  // D# group 1: mask=0, data_size=1(2B), dims/strides for 2D 64x128 tile
  i32x8 g1;
  g1[0] = (1 << 16);          // workgroup_mask=0, data_size=2 bytes, no flags
  g1[1] = (64 << 16);         // atomic_barrier=0 | tensor_dim0[15:0]=64
  g1[2] = (int)(8192u << 16); // tensor_dim0[31:16]=0 | tensor_dim1[15:0]=8192
  g1[3] = (64 << 16);         // tensor_dim1[31:16]=0 | tile_dim0=64
  g1[4] = 128;                // tile_dim1=128 | tile_dim2=0
  g1[5] = 64;                 // tensor_dim0_stride[31:0]=64
  g1[6] = 0;                  // stride hi | tensor_dim1_stride lo
  g1[7] = 0;
  i32x4 gz = {0, 0, 0, 0};    // groups 2/3 unused (2-D tensor)
#if __clang_major__ >= 23
  i32x8 gz8 = {0, 0, 0, 0, 0, 0, 0, 0};
  __builtin_amdgcn_tensor_load_to_lds(g0, g1, gz, gz, gz8, 0);
#else
  __builtin_amdgcn_tensor_load_to_lds(g0, g1, gz, gz, 0);
#endif
}

__global__ __launch_bounds__(256) void adj_kernel(const _Float16* __restrict__ Vh,
                                                  float* __restrict__ adj) {
  __shared__ __align__(16) _Float16 colT[128 * 64];
  const int wave = threadIdx.x >> 5;
  const int lane = threadIdx.x & 31;
  const int rowbase = blockIdx.y * 128 + wave * 16;
  const int colbase0 = blockIdx.x * 128;
  if (wave == 0) {
    tdm_load_tile_128x64(Vh + (size_t)colbase0 * 64, colT);
  }
  // A fragments (own 16 rows, contiguous 16B f16 loads) overlap the DMA.
  const _Float16* ap = Vh + (size_t)(rowbase + (lane & 15)) * 64 + ((lane >> 4) << 3);
  v8h alo0 = *(const v8h*)(ap);
  v8h ahi0 = *(const v8h*)(ap + 16);
  v8h alo1 = *(const v8h*)(ap + 32);
  v8h ahi1 = *(const v8h*)(ap + 48);
  v16h a0, a1;
#pragma unroll
  for (int i = 0; i < 8; ++i) {
    a0[i] = alo0[i]; a0[8 + i] = ahi0[i];
    a1[i] = alo1[i]; a1[8 + i] = ahi1[i];
  }
  if (wave == 0) {
    __builtin_amdgcn_s_wait_tensorcnt(0);
  }
  __syncthreads();
#pragma unroll
  for (int ct = 0; ct < 8; ++ct) {
    v16h b0 = load_bfrag_lds(colT, ct * 16, 0);
    v16h b1 = load_bfrag_lds(colT, ct * 16, 32);
    v8f acc = {};
    acc = __builtin_amdgcn_wmma_f32_16x16x32_f16(false, a0, false, b0,
                                                 (short)0, acc, false, false);
    acc = __builtin_amdgcn_wmma_f32_16x16x32_f16(false, a1, false, b1,
                                                 (short)0, acc, false, false);
    const int c = colbase0 + ct * 16 + (lane & 15);
#pragma unroll
    for (int j = 0; j < 8; ++j) {
      const int rr = rowbase + j + ((lane >> 4) << 3);
      adj[(size_t)rr * NNODES + c] = 1.f / (1.f + __expf(-acc[j]));
    }
  }
}

// ---------------------------------------------------------------------------
// Sparse aggregation: out[dst] += w[e] * h[src], edge-parallel f32 atomics.
// ---------------------------------------------------------------------------
__global__ void zero_kernel(float* __restrict__ p, int n) {
  int i = blockIdx.x * blockDim.x + threadIdx.x;
  if (i < n) p[i] = 0.f;
}

__global__ void spmm_kernel(const float* __restrict__ h, const int* __restrict__ src,
                            const int* __restrict__ dst, const float* __restrict__ w,
                            float* __restrict__ out) {
  int t = blockIdx.x * blockDim.x + threadIdx.x;
  int e = t >> 2;
  if (e >= NEDGE) return;
  int c0 = (t & 3) << 4;
  float ww = w[e];
  const float* hp = h + (size_t)src[e] * 64 + c0;
  float* op = out + (size_t)dst[e] * 64 + c0;
#pragma unroll
  for (int c = 0; c < 16; ++c) atomicAdd(op + c, ww * hp[c]);
}

__global__ void bias_prelu_kernel(const float* __restrict__ agg,
                                  const float* __restrict__ bias,
                                  const float* __restrict__ slope,
                                  float* __restrict__ out) {
  int i = blockIdx.x * blockDim.x + threadIdx.x;
  if (i >= N64) return;
  float v = agg[i] + bias[i & 63];
  float sl = *slope;
  out[i] = (v >= 0.f) ? v : sl * v;
}

__global__ void cvt_h_kernel(const float* __restrict__ x, _Float16* __restrict__ y) {
  int i = blockIdx.x * blockDim.x + threadIdx.x;
  if (i < N64) y[i] = (_Float16)x[i];
}

// ---------------------------------------------------------------------------
// Bilinear discriminators: pout[n] = <rows[n], keep[n]> + bd,
// nout[k,n] = <rows[neg[k,n]], keep[n]> + bd.
// ---------------------------------------------------------------------------
__device__ __forceinline__ float dot64(const float* __restrict__ a,
                                       const float4* __restrict__ b16) {
  const float4* a4 = (const float4*)a;
  float acc = 0.f;
#pragma unroll
  for (int i = 0; i < 16; ++i) {
    float4 u = a4[i], v = b16[i];
    acc += u.x * v.x + u.y * v.y + u.z * v.z + u.w * v.w;
  }
  return acc;
}

__global__ void disc_kernel(const float* __restrict__ rows, const float* __restrict__ keep,
                            const int* __restrict__ neg, const float* __restrict__ bdp,
                            float* __restrict__ pout, float* __restrict__ nout) {
  int n = blockIdx.x * blockDim.x + threadIdx.x;
  if (n >= NNODES) return;
  float4 kv[16];
  const float4* k4 = (const float4*)(keep + (size_t)n * 64);
#pragma unroll
  for (int i = 0; i < 16; ++i) kv[i] = k4[i];
  const float bd = *bdp;
  pout[n] = dot64(rows + (size_t)n * 64, kv) + bd;
#pragma unroll
  for (int k = 0; k < NNEG; ++k) {
    int m = neg[k * NNODES + n];
    nout[k * NNODES + n] = dot64(rows + (size_t)m * 64, kv) + bd;
  }
}

__global__ void score_kernel(const float* __restrict__ nv, const float* __restrict__ Ws,
                             const float* __restrict__ bsp, float* __restrict__ score) {
  int n = blockIdx.x * blockDim.x + threadIdx.x;
  if (n >= NNODES) return;
  float acc = *bsp;
  const float* p = nv + (size_t)n * 64;
#pragma unroll
  for (int t = 0; t < 64; ++t) acc += p[t] * Ws[t];
  score[n] = 1.f / (1.f + __expf(-acc));
}

// ---------------------------------------------------------------------------
// Per-subgraph: hom quadratic forms + max/sum stats (sort eliminated:
// the mean is sort-invariant; a_sort[0]=max; a_at_avg = max of values below
// the mean, falling back to max when no element is below it).
// ---------------------------------------------------------------------------
__device__ __forceinline__ float block_reduce(float v, float* red, int domax) {
  int k = threadIdx.x;
  red[k] = v;
  __syncthreads();
  for (int off = 64; off > 0; off >>= 1) {
    if (k < off) red[k] = domax ? fmaxf(red[k], red[k + off]) : red[k] + red[k + off];
    __syncthreads();
  }
  float r = red[0];
  __syncthreads();
  return r;
}

__global__ __launch_bounds__(128) void subg_kernel(
    const float* __restrict__ score, const int* __restrict__ a_idx,
    const int* __restrict__ n_idx, const float* __restrict__ a_lap,
    const float* __restrict__ n_lap, float* __restrict__ partials) {
  __shared__ float aS[KSUB], nS[KSUB], red[KSUB];
  const int s = blockIdx.x, k = threadIdx.x;
  aS[k] = score[a_idx[s * KSUB + k]];
  nS[k] = score[n_idx[s * KSUB + k]];
  __syncthreads();
  const float* al = a_lap + ((size_t)s * KSUB + k) * KSUB;
  const float* nl = n_lap + ((size_t)s * KSUB + k) * KSUB;
  float ia = 0.f, in = 0.f;
  for (int j = 0; j < KSUB; ++j) { ia += al[j] * aS[j]; in += nl[j] * nS[j]; }
  float ak = aS[k], nk = nS[k];
  float homs = block_reduce(ak * ia + nk * in, red, 0);
  float aMax = block_reduce(ak, red, 1);
  float nMax = block_reduce(nk, red, 1);
  float sumA = block_reduce(ak, red, 0);
  float meanA = sumA * (1.f / (float)KSUB);
  float below = block_reduce((ak < meanA) ? ak : -3.0e38f, red, 1);
  if (k == 0) {
    float a_at_avg = (below > -1.0e38f) ? below : aMax;
    partials[s * 4 + 0] = fmaxf(0.f, 1.f - aMax + nMax);
    partials[s * 4 + 1] = fmaxf(0.f, 1.f - aMax + a_at_avg);
    partials[s * 4 + 2] = sumA;
    partials[s * 4 + 3] = homs;
  }
}

__global__ __launch_bounds__(128) void final_kernel(const float* __restrict__ partials,
                                                    float* __restrict__ out) {
  __shared__ float red[NSUB];
  int k = threadIdx.x;
  float s0 = block_reduce(partials[k * 4 + 0], red, 0);
  float s1 = block_reduce(partials[k * 4 + 1], red, 0);
  float s2 = block_reduce(partials[k * 4 + 2], red, 0);
  float s3 = block_reduce(partials[k * 4 + 3], red, 0);
  if (k == 0) {
    out[0] = s0 / (float)NSUB;
    out[1] = s1 / (float)NSUB;
    out[2] = s2 / (float)NSUB;
    out[3] = s3 / (2.f * (float)NSUB);
  }
}

// ---------------------------------------------------------------------------
extern "C" void kernel_launch(void* const* d_in, const int* in_sizes, int n_in,
                              void* d_out, int out_size, void* d_ws, size_t ws_size,
                              hipStream_t stream) {
  (void)in_sizes; (void)n_in; (void)out_size; (void)ws_size;
  const float* feat = (const float*)d_in[0];
  const int*   esrc = (const int*)d_in[1];
  const int*   edst = (const int*)d_in[2];
  const float* ew   = (const float*)d_in[3];
  const float* aw   = (const float*)d_in[4];
  const int*   neg1 = (const int*)d_in[5];
  const int*   neg2 = (const int*)d_in[6];
  const int*   aidx = (const int*)d_in[7];
  const int*   nidx = (const int*)d_in[8];
  const float* alap = (const float*)d_in[9];
  const float* nlap = (const float*)d_in[10];
  const float* W1 = (const float*)d_in[11]; const float* b1 = (const float*)d_in[12];
  const float* W2 = (const float*)d_in[13]; const float* b2 = (const float*)d_in[14];
  const float* Wq = (const float*)d_in[15]; const float* bq = (const float*)d_in[16];
  const float* Ws = (const float*)d_in[17]; const float* bs = (const float*)d_in[18];
  const float* B1 = (const float*)d_in[19]; const float* bd1 = (const float*)d_in[20];
  const float* B2 = (const float*)d_in[21]; const float* bd2 = (const float*)d_in[22];
  const float* a1 = (const float*)d_in[23]; const float* a2 = (const float*)d_in[24];
  const float* aq = (const float*)d_in[25];

  float* out   = (float*)d_out;
  float* o_adj = out + 4;
  float* o_p1  = out + 4 + (size_t)NNODES * NNODES;
  float* o_n1  = o_p1 + NNODES;
  float* o_p2  = o_n1 + NNEG * NNODES;
  float* o_n2  = o_p2 + NNODES;

  float* ws = (float*)d_ws;
  float* h1 = ws + 0 * (size_t)N64;
  float* v1 = ws + 1 * (size_t)N64;
  float* h2 = ws + 2 * (size_t)N64;
  float* v2 = ws + 3 * (size_t)N64;
  float* nv = ws + 4 * (size_t)N64;
  float* nn = ws + 5 * (size_t)N64;
  float* u1 = ws + 6 * (size_t)N64;
  float* y2 = ws + 7 * (size_t)N64;
  float* agg = ws + 8 * (size_t)N64;
  float* score = ws + 9 * (size_t)N64;
  float* partials = score + NNODES;
  _Float16* Vh = (_Float16*)(ws + 10 * (size_t)N64);

  dim3 b256(256);
  dim3 ggemm(NNODES / 128);
  int gelt = (N64 + 255) / 256;
  int gedge = (NEDGE * 4) / 256;
  int gnode = (NNODES + 255) / 256;

  // GCN layer 1 pre-propagation + discriminator projection of feat
  gemm64_kernel<<<ggemm, b256, 0, stream>>>(feat, W1, nullptr, nullptr, h1, 0, 0);
  gemm64_kernel<<<ggemm, b256, 0, stream>>>(feat, B1, nullptr, nullptr, u1, 0, 0);
  // agg1 -> v1
  zero_kernel<<<gelt, b256, 0, stream>>>(agg, N64);
  spmm_kernel<<<gedge, b256, 0, stream>>>(h1, esrc, edst, ew, agg);
  bias_prelu_kernel<<<gelt, b256, 0, stream>>>(agg, b1, a1, v1);
  // node_neighbor = agg(h1, avg_w)
  zero_kernel<<<gelt, b256, 0, stream>>>(nn, N64);
  spmm_kernel<<<gedge, b256, 0, stream>>>(h1, esrc, edst, aw, nn);
  // GCN layer 2
  gemm64_kernel<<<ggemm, b256, 0, stream>>>(v1, W2, nullptr, nullptr, h2, 0, 0);
  zero_kernel<<<gelt, b256, 0, stream>>>(agg, N64);
  spmm_kernel<<<gedge, b256, 0, stream>>>(h2, esrc, edst, ew, agg);
  bias_prelu_kernel<<<gelt, b256, 0, stream>>>(agg, b2, a2, v2);
  // node_vec = prelu(v2 @ Wq + bq)
  gemm64_kernel<<<ggemm, b256, 0, stream>>>(v2, Wq, bq, aq, nv, 0, 1);
  // y2 = node_neighbor @ B2^T
  gemm64_kernel<<<ggemm, b256, 0, stream>>>(nn, B2, nullptr, nullptr, y2, 1, 0);
  // adj_rebuilt (dominant: 256MB store; TDM-staged WMMA GEMM + fused sigmoid)
  cvt_h_kernel<<<gelt, b256, 0, stream>>>(nv, Vh);
  dim3 gadj(NNODES / 128, NNODES / 128);
  adj_kernel<<<gadj, b256, 0, stream>>>(Vh, o_adj);
  // discriminators
  disc_kernel<<<gnode, b256, 0, stream>>>(u1, nv, neg1, bd1, o_p1, o_n1);
  disc_kernel<<<gnode, b256, 0, stream>>>(nv, y2, neg2, bd2, o_p2, o_n2);
  // MIL scores + subgraph losses
  score_kernel<<<gnode, b256, 0, stream>>>(nv, Ws, bs, score);
  subg_kernel<<<NSUB, 128, 0, stream>>>(score, aidx, nidx, alap, nlap, partials);
  final_kernel<<<1, 128, 0, stream>>>(partials, out);
}
